// LSTM_attn_84731114816303
// MI455X (gfx1250) — compile-verified, matrix operand order checked by reading, not run on hardware
//
#include <hip/hip_runtime.h>
#include <hip/hip_bf16.h>

// ---------------------------------------------------------------------------
// B=64, S=512, E=256, H=512, O=256
//   x  : (S*B=32768, 2E=512)      W_ih/W_hh : (4H=2048, 512)
//   xg : (2, 32768, 2048) bf16    out       : (B, S, 2H=1024) bf16
// ---------------------------------------------------------------------------
#define BB   64
#define SS   512
#define KIN  512
#define HH   512
#define G4   2048
#define DD   1024
#define NBLK 16

typedef __attribute__((ext_vector_type(16))) __bf16        v16bf;
typedef __attribute__((ext_vector_type(8)))  float         v8f;
typedef __attribute__((ext_vector_type(4)))  unsigned int  v4ui;
typedef __attribute__((ext_vector_type(8)))  int           v8i;
typedef __attribute__((ext_vector_type(4)))  int           v4i;

#ifndef HAVE_TDM
# if defined(__has_builtin)
#  if __has_builtin(__builtin_amdgcn_tensor_load_to_lds) && __has_builtin(__builtin_amdgcn_s_wait_tensorcnt)
#   define HAVE_TDM 1
#  else
#   define HAVE_TDM 0
#  endif
# else
#  define HAVE_TDM 0
# endif
#endif

union TileU { uint4 q[2]; v16bf v; };

__device__ inline unsigned short f2bf(float f) {
  unsigned u = __float_as_uint(f);
  return (unsigned short)((u + 0x7FFFu + ((u >> 16) & 1u)) >> 16);
}
__device__ inline float bf2f(unsigned short h) {
  return __uint_as_float(((unsigned)h) << 16);
}

// 16x32 bf16 tile, row-major [row][k], into WMMA per-lane layout.
__device__ inline v16bf load_tile16x32(const unsigned short* __restrict__ base, int ld) {
  int lane = threadIdx.x & 31;
  int row  = lane & 15;
  int kh   = lane >> 4;
  const unsigned short* p = base + (size_t)row * ld + kh * 8;
  TileU u;
  u.q[0] = *(const uint4*)(p);
  u.q[1] = *(const uint4*)(p + 16);
  return u.v;
}

__device__ inline v8f wmma_bf16(v16bf a, v16bf b, v8f c) {
  return __builtin_amdgcn_wmma_f32_16x16x32_bf16(false, a, false, b, (short)0, c, false, false);
}

#if HAVE_TDM
// Tensor Data Mover: DMA a (tile_d0 x tile_d1 x tile_d2) tile of 2-byte
// elements from global into LDS (packed x-fastest). Strides in elements.
// D# layout per CDNA5 ISA.
__device__ inline void tdm_load_to_lds(void* lds_ptr, const void* gptr,
    unsigned tile_d0, unsigned tile_d1, unsigned tile_d2,
    unsigned tensor_d0, unsigned tensor_d1, unsigned tensor_d2,
    unsigned long long stride0, unsigned long long stride1)
{
  unsigned lds_off = (unsigned)(unsigned long long)(uintptr_t)lds_ptr; // flat[31:0] == LDS offset
  unsigned long long ga = (unsigned long long)(uintptr_t)gptr;
  v4ui g0;
  g0.x = 1u;                                                  // count=1 (valid), user mode
  g0.y = lds_off;                                             // lds_addr
  g0.z = (unsigned)(ga & 0xFFFFFFFFull);                      // global_addr[31:0]
  g0.w = (unsigned)((ga >> 32) & 0x01FFFFFFull) | (2u << 30); // global_addr[56:32] | type=2
  v8i g1;
  g1[0] = (int)(1u << 16);                                    // data_size=1 (2B); mask=0; no flags
  g1[1] = (int)((tensor_d0 & 0xFFFFu) << 16);                 // tensor_dim0[15:0]
  g1[2] = (int)(((tensor_d0 >> 16) & 0xFFFFu) | ((tensor_d1 & 0xFFFFu) << 16));
  g1[3] = (int)(((tensor_d1 >> 16) & 0xFFFFu) | ((tile_d0 & 0xFFFFu) << 16));
  g1[4] = (int)((tile_d1 & 0xFFFFu) | ((tile_d2 & 0xFFFFu) << 16));
  g1[5] = (int)(stride0 & 0xFFFFFFFFull);                     // tensor_dim0_stride[31:0]
  g1[6] = (int)(((stride0 >> 32) & 0xFFFFull) | ((stride1 & 0xFFFFull) << 16));
  g1[7] = (int)((stride1 >> 16) & 0xFFFFFFFFull);
  v4i g2; g2[0] = (int)tensor_d2; g2[1] = 0; g2[2] = 0; g2[3] = 0;
  v4i g3; g3[0] = 0; g3[1] = 0; g3[2] = 0; g3[3] = 0;
#if __clang_major__ >= 23
  v8i g4 = {0, 0, 0, 0, 0, 0, 0, 0};
  __builtin_amdgcn_tensor_load_to_lds(g0, g1, g2, g3, g4, 0);
#else
  __builtin_amdgcn_tensor_load_to_lds(g0, g1, g2, g3, 0);
#endif
}
#endif

// ---------------------------------------------------------------------------
// Prep kernels
// ---------------------------------------------------------------------------
__global__ __launch_bounds__(256) void k_f2bf(const float* __restrict__ src,
                                              unsigned short* __restrict__ dst, int n) {
  int i = blockIdx.x * 256 + threadIdx.x;
  if (i < n) dst[i] = f2bf(src[i]);
}

__global__ __launch_bounds__(256) void k_transpose_x(const float* __restrict__ in,
                                                     unsigned short* __restrict__ x_bf) {
  long i = (long)blockIdx.x * 256 + threadIdx.x;
  int  k = (int)(i & 511);
  long r = i >> 9;
  int  b = (int)(r & 63);
  int  s = (int)(r >> 6);
  x_bf[i] = f2bf(in[(long)b * (SS * KIN) + (long)s * KIN + k]);
}

__global__ __launch_bounds__(256) void k_bias(const float* bif, const float* bhf,
                                              const float* bib, const float* bhb,
                                              float* __restrict__ bias) {
  int i = blockIdx.x * 256 + threadIdx.x;
  if (i < G4)          bias[i] = bif[i] + bhf[i];
  else if (i < 2 * G4) bias[i] = bib[i - G4] + bhb[i - G4];
}

__global__ __launch_bounds__(256) void k_zero_u32(unsigned* __restrict__ p, int n) {
  int i = blockIdx.x * 256 + threadIdx.x;
  if (i < n) p[i] = 0u;
}

// ---------------------------------------------------------------------------
// Input GEMM: xg[dir][m][n] = sum_k x[m][k]*W_ih[dir][n][k] + bias[dir][n]
// grid (1024 strips of 32 rows, 4, 2), 256 thr.
// Block stages its 32x512 A strip in LDS via TDM; each wave does
// 2 M-tiles x 4 N-tiles. k-loop manually unrolled x2 with ping-pong
// B-register sets (no copy chains).
// ---------------------------------------------------------------------------
__global__ __launch_bounds__(256) void k_gemm_in(
    const unsigned short* __restrict__ x_bf,   // [32768][512]
    const unsigned short* __restrict__ Wih,    // [2][2048][512]
    const float* __restrict__ bias,            // [2][2048]
    unsigned short* __restrict__ xg)           // [2][32768][2048]
{
  const int wave = threadIdx.x >> 5, lane = threadIdx.x & 31;
  const int mb  = blockIdx.x * 32;
  const int nb  = (blockIdx.y * 8 + wave) * 64;
  const int dir = blockIdx.z;
  const unsigned short* W = Wih + (size_t)dir * G4 * KIN;

  __shared__ unsigned short sA[32 * KIN];      // 32 KB A strip

#if HAVE_TDM
  if (threadIdx.x < 32) {
    tdm_load_to_lds(sA, x_bf + (size_t)mb * KIN,
                    /*tile*/ 32 * KIN, 1, 0,
                    /*tensor*/ 32 * KIN, 1, 0,
                    /*strides*/ 32 * KIN, 0);
    __builtin_amdgcn_s_wait_tensorcnt(0);
  }
#else
  for (int i = threadIdx.x * 8; i < 32 * KIN; i += 256 * 8)
    *(uint4*)&sA[i] = *(const uint4*)&x_bf[(size_t)mb * KIN + i];
#endif
  __syncthreads();

  v8f acc[2][4];
#pragma unroll
  for (int m = 0; m < 2; ++m)
#pragma unroll
    for (int j = 0; j < 4; ++j) acc[m][j] = (v8f){};

  v16bf bp[4], bq[4];                          // ping-pong B tile sets
#pragma unroll
  for (int j = 0; j < 4; ++j)
    bp[j] = load_tile16x32(W + (size_t)(nb + j * 16) * KIN, KIN);

  for (int k0 = 0; k0 < KIN; k0 += 64) {
    const int ka = k0 + 32, kb = k0 + 64;
    // prefetch set q @ ka (always valid: KIN % 64 == 0)
#pragma unroll
    for (int j = 0; j < 4; ++j)
      bq[j] = load_tile16x32(W + (size_t)(nb + j * 16) * KIN + ka, KIN);
    {
      v16bf a0 = load_tile16x32(sA + k0, KIN);            // ds_load
      v16bf a1 = load_tile16x32(sA + 16 * KIN + k0, KIN); // ds_load
#pragma unroll
      for (int j = 0; j < 4; ++j) acc[0][j] = wmma_bf16(a0, bp[j], acc[0][j]);
#pragma unroll
      for (int j = 0; j < 4; ++j) acc[1][j] = wmma_bf16(a1, bp[j], acc[1][j]);
    }
    // prefetch set p @ kb
    if (kb < KIN) {
#pragma unroll
      for (int j = 0; j < 4; ++j)
        bp[j] = load_tile16x32(W + (size_t)(nb + j * 16) * KIN + kb, KIN);
    }
    {
      v16bf a0 = load_tile16x32(sA + ka, KIN);
      v16bf a1 = load_tile16x32(sA + 16 * KIN + ka, KIN);
#pragma unroll
      for (int j = 0; j < 4; ++j) acc[0][j] = wmma_bf16(a0, bq[j], acc[0][j]);
#pragma unroll
      for (int j = 0; j < 4; ++j) acc[1][j] = wmma_bf16(a1, bq[j], acc[1][j]);
    }
  }

  unsigned short* outp = xg + (size_t)dir * (SS * (size_t)BB) * G4;
  const float* bs = bias + dir * G4;
  const int col = lane & 15, rowoff = (lane >> 4) * 8;
#pragma unroll
  for (int m = 0; m < 2; ++m)
#pragma unroll
    for (int j = 0; j < 4; ++j) {
      int n = nb + j * 16 + col;
      float bv = bs[n];
#pragma unroll
      for (int r = 0; r < 8; ++r) {
        int row = mb + m * 16 + rowoff + r;
        outp[(size_t)row * G4 + n] = f2bf(acc[m][j][r] + bv);
      }
    }
}

// ---------------------------------------------------------------------------
// Recurrent scan (persistent). grid (NBLK, 2 dirs), 256 thr (8 waves).
// Per step: TDM-stage the xg gate slice (32x4x64 tile) into LDS while the
// WMMA k-loop runs; h double-buffered in global + device-scope barrier.
// k-loop manually unrolled x2 with ping-pong register sets.
// ---------------------------------------------------------------------------
__global__ __launch_bounds__(256) void k_lstm_scan(
    const unsigned short* __restrict__ xg,     // [2][32768][2048]
    const unsigned short* __restrict__ Whh,    // [2][2048][512]
    unsigned short* __restrict__ h_glob,       // [2 buf][2 dir][64][512]
    unsigned short* __restrict__ out_bf,       // [64][512][1024]
    unsigned int* __restrict__ barp)           // [2]
{
  const int tid = threadIdx.x, wave = tid >> 5, lane = tid & 31;
  const int blk = blockIdx.x, dir = blockIdx.y;
  const int j0 = blk * 32;
  const unsigned short* W   = Whh + (size_t)dir * G4 * HH;
  const unsigned short* xgd = xg  + (size_t)dir * (SS * (size_t)BB) * G4;
  const int gate  = wave >> 1;
  const int nbase = gate * HH + j0 + (wave & 1) * 16;
  const int col = lane & 15, rowoff = (lane >> 4) * 8;

  __shared__ float          g_lds[BB * 128];   // 32 KB
  __shared__ float          c_lds[BB * 32];    //  8 KB
  __shared__ unsigned short xg_s[BB * 128];    // 16 KB: [b][gate][jj]
  for (int i = tid; i < BB * 32; i += 256) c_lds[i] = 0.f;
  __syncthreads();

  for (int step = 0; step < SS; ++step) {
    const int t = (dir == 0) ? step : (SS - 1 - step);
    const unsigned short* hr = h_glob + ((size_t)(step & 1) * 2 + dir) * (BB * HH);
    unsigned short*       hw = h_glob + ((size_t)((step + 1) & 1) * 2 + dir) * (BB * HH);

    // stage this step's xg gate slice; overlaps the WMMA loop below
#if HAVE_TDM
    if (tid < 32) {
      tdm_load_to_lds(xg_s, xgd + (size_t)t * BB * G4 + j0,
                      /*tile*/ 32, 4, BB,
                      /*tensor*/ 32, 4, BB,
                      /*strides*/ HH, G4);
    }
#else
    for (int i = tid; i < BB * 128; i += 256) {
      int jj = i & 31, g = (i >> 5) & 3, b = i >> 7;
      xg_s[i] = xgd[((size_t)t * BB + b) * G4 + g * HH + j0 + jj];
    }
#endif

    // h @ W_hh^T for this block's 8 N-tiles, ping-pong pipelined loads
    v8f acc0 = {}, acc1 = {}, acc2 = {}, acc3 = {};
    const unsigned short* Wb = W + (size_t)nbase * HH;
    v16bf pb0  = load_tile16x32(Wb, HH);
    v16bf pa00 = load_tile16x32(hr, HH);
    v16bf pa01 = load_tile16x32(hr + 16 * HH, HH);
    v16bf pa02 = load_tile16x32(hr + 32 * HH, HH);
    v16bf pa03 = load_tile16x32(hr + 48 * HH, HH);
    for (int k0 = 0; k0 < HH; k0 += 64) {
      const int ka = k0 + 32, kb = k0 + 64;
      // prefetch set 1 @ ka (always valid: HH % 64 == 0)
      v16bf pb1  = load_tile16x32(Wb + ka, HH);
      v16bf pa10 = load_tile16x32(hr + ka, HH);
      v16bf pa11 = load_tile16x32(hr + 16 * HH + ka, HH);
      v16bf pa12 = load_tile16x32(hr + 32 * HH + ka, HH);
      v16bf pa13 = load_tile16x32(hr + 48 * HH + ka, HH);
      acc0 = wmma_bf16(pa00, pb0, acc0);
      acc1 = wmma_bf16(pa01, pb0, acc1);
      acc2 = wmma_bf16(pa02, pb0, acc2);
      acc3 = wmma_bf16(pa03, pb0, acc3);
      // prefetch set 0 @ kb
      if (kb < HH) {
        pb0  = load_tile16x32(Wb + kb, HH);
        pa00 = load_tile16x32(hr + kb, HH);
        pa01 = load_tile16x32(hr + 16 * HH + kb, HH);
        pa02 = load_tile16x32(hr + 32 * HH + kb, HH);
        pa03 = load_tile16x32(hr + 48 * HH + kb, HH);
      }
      acc0 = wmma_bf16(pa10, pb1, acc0);
      acc1 = wmma_bf16(pa11, pb1, acc1);
      acc2 = wmma_bf16(pa12, pb1, acc2);
      acc3 = wmma_bf16(pa13, pb1, acc3);
    }

    // scatter accumulators to LDS gate buffer [b][gate*32 + jj]
    {
      v8f* accs[4] = {&acc0, &acc1, &acc2, &acc3};
#pragma unroll
      for (int m = 0; m < 4; ++m) {
        v8f& A = *accs[m];
#pragma unroll
        for (int r = 0; r < 8; ++r) {
          int row = m * 16 + rowoff + r;
          g_lds[row * 128 + wave * 16 + col] = A[r];
        }
      }
    }
#if HAVE_TDM
    if (tid < 32) __builtin_amdgcn_s_wait_tensorcnt(0);
#endif
    __syncthreads();

    // elementwise LSTM cell: 2048 elems / 256 thr = 8 each
    for (int i = 0; i < 8; ++i) {
      int e  = tid + i * 256;
      int b  = e >> 5, jj = e & 31;
      float ip = g_lds[b * 128 +  0 + jj] + bf2f(xg_s[(b * 4 + 0) * 32 + jj]);
      float fp = g_lds[b * 128 + 32 + jj] + bf2f(xg_s[(b * 4 + 1) * 32 + jj]);
      float gp = g_lds[b * 128 + 64 + jj] + bf2f(xg_s[(b * 4 + 2) * 32 + jj]);
      float op = g_lds[b * 128 + 96 + jj] + bf2f(xg_s[(b * 4 + 3) * 32 + jj]);
      float iv = 1.f / (1.f + __expf(-ip));
      float fv = 1.f / (1.f + __expf(-fp));
      float gv = tanhf(gp);
      float ov = 1.f / (1.f + __expf(-op));
      float c  = fv * c_lds[e] + iv * gv;
      c_lds[e] = c;
      unsigned short hb16 = f2bf(ov * tanhf(c));
      hw[b * HH + j0 + jj] = hb16;
      out_bf[((size_t)b * SS + t) * DD + dir * HH + j0 + jj] = hb16;
    }

    // device-scope step barrier (per direction)
    __threadfence();
    __syncthreads();
    if (tid == 0) {
      __hip_atomic_fetch_add(&barp[dir], 1u, __ATOMIC_ACQ_REL, __HIP_MEMORY_SCOPE_AGENT);
      unsigned target = (unsigned)(step + 1) * NBLK;
      while (__hip_atomic_load(&barp[dir], __ATOMIC_ACQUIRE, __HIP_MEMORY_SCOPE_AGENT) < target) {
        __builtin_amdgcn_s_sleep(1);
      }
    }
    __syncthreads();
    __threadfence();
  }
}

// ---------------------------------------------------------------------------
// Attention (only the last query row is consumed) + output projection.
// ---------------------------------------------------------------------------
__global__ __launch_bounds__(256) void k_attn_out(
    const unsigned short* __restrict__ out_bf, // [64][512][1024]
    const float* __restrict__ W_out,           // [256][1024]
    const float* __restrict__ b_out,           // [256]
    float* __restrict__ y)                     // [64][256]
{
  const int b = blockIdx.x, tid = threadIdx.x;
  __shared__ float q_s[DD];
  __shared__ float w_s[SS];
  __shared__ float red[256];
  __shared__ float ctx[DD];
  const unsigned short* ob = out_bf + (size_t)b * SS * DD;

  for (int d = tid; d < DD; d += 256) q_s[d] = bf2f(ob[(size_t)(SS - 1) * DD + d]);
  __syncthreads();

  for (int t = tid; t < SS; t += 256) {
    const unsigned short* row = ob + (size_t)t * DD;
    float s = 0.f;
    for (int d = 0; d < DD; ++d) s += q_s[d] * bf2f(row[d]);
    s *= (1.0f / 32.0f);
    w_s[t] = (s > 0.1f) ? s : -1.0e6f;
  }
  __syncthreads();

  float lmax = fmaxf(w_s[tid], w_s[tid + 256]);
  red[tid] = lmax; __syncthreads();
  for (int off = 128; off > 0; off >>= 1) {
    if (tid < off) red[tid] = fmaxf(red[tid], red[tid + off]);
    __syncthreads();
  }
  float mx = red[0];
  __syncthreads();

  float lsum = 0.f;
  for (int t = tid; t < SS; t += 256) {
    float e = __expf(w_s[t] - mx);
    w_s[t] = e; lsum += e;
  }
  red[tid] = lsum; __syncthreads();
  for (int off = 128; off > 0; off >>= 1) {
    if (tid < off) red[tid] += red[tid + off];
    __syncthreads();
  }
  float inv = 1.0f / red[0];
  __syncthreads();

  for (int d = tid; d < DD; d += 256) {
    float a = 0.f;
    for (int t = 0; t < SS; ++t) a += w_s[t] * bf2f(ob[(size_t)t * DD + d]);
    ctx[d] = a * inv;
  }
  __syncthreads();

  float a = b_out[tid];
  const float* wr = W_out + (size_t)tid * DD;
  for (int d = 0; d < DD; ++d) a += wr[d] * ctx[d];
  y[(size_t)b * 256 + tid] = a;
}

// ---------------------------------------------------------------------------
// Host launcher
// ---------------------------------------------------------------------------
extern "C" void kernel_launch(void* const* d_in, const int* in_sizes, int n_in,
                              void* d_out, int out_size, void* d_ws, size_t ws_size,
                              hipStream_t stream) {
  (void)in_sizes; (void)n_in; (void)out_size; (void)ws_size;

  const float* inputs = (const float*)d_in[0];
  const float* W_ih_f = (const float*)d_in[1];
  const float* W_hh_f = (const float*)d_in[2];
  const float* b_ih_f = (const float*)d_in[3];
  const float* b_hh_f = (const float*)d_in[4];
  const float* W_ih_b = (const float*)d_in[5];
  const float* W_hh_b = (const float*)d_in[6];
  const float* b_ih_b = (const float*)d_in[7];
  const float* b_hh_b = (const float*)d_in[8];
  const float* W_out  = (const float*)d_in[9];
  const float* b_out  = (const float*)d_in[10];

  char* base = (char*)d_ws;
  size_t off = 0;
  auto take = [&](size_t bytes) -> char* {
    char* p = base + off;
    off = (off + bytes + 255) & ~(size_t)255;
    return p;
  };

  unsigned short* x_bf   = (unsigned short*)take((size_t)SS * BB * KIN * 2);
  unsigned short* Wih_bf = (unsigned short*)take((size_t)2 * G4 * KIN * 2);
  unsigned short* Whh_bf = (unsigned short*)take((size_t)2 * G4 * HH * 2);
  float*          bias   = (float*)take((size_t)2 * G4 * 4);
  unsigned short* xg_bf  = (unsigned short*)take((size_t)2 * SS * BB * G4 * 2);
  unsigned short* out_bf = (unsigned short*)take((size_t)BB * SS * DD * 2);
  unsigned short* h_glob = (unsigned short*)take((size_t)2 * 2 * BB * HH * 2);
  unsigned int*   barp   = (unsigned int*)take(256);

  const int WN = G4 * KIN;
  k_f2bf<<<(WN + 255) / 256, 256, 0, stream>>>(W_ih_f, Wih_bf,      WN);
  k_f2bf<<<(WN + 255) / 256, 256, 0, stream>>>(W_ih_b, Wih_bf + WN, WN);
  k_f2bf<<<(WN + 255) / 256, 256, 0, stream>>>(W_hh_f, Whh_bf,      WN);
  k_f2bf<<<(WN + 255) / 256, 256, 0, stream>>>(W_hh_b, Whh_bf + WN, WN);
  k_bias<<<16, 256, 0, stream>>>(b_ih_f, b_hh_f, b_ih_b, b_hh_b, bias);
  k_zero_u32<<<(2 * 2 * BB * HH * 2 / 4 + 255) / 256, 256, 0, stream>>>((unsigned*)h_glob, 2 * 2 * BB * HH * 2 / 4);
  k_zero_u32<<<1, 256, 0, stream>>>(barp, 64);
  k_transpose_x<<<((size_t)SS * BB * KIN) / 256, 256, 0, stream>>>(inputs, x_bf);

  // input GEMM: 32-row strips
  k_gemm_in<<<dim3((SS * BB) / 32, 4, 2), 256, 0, stream>>>(x_bf, Wih_bf, bias, xg_bf);

  // recurrent scan (persistent)
  k_lstm_scan<<<dim3(NBLK, 2), 256, 0, stream>>>(xg_bf, Whh_bf, h_glob, out_bf, barp);

  // last-row attention + output projection
  k_attn_out<<<BB, 256, 0, stream>>>(out_bf, W_out, b_out, (float*)d_out);
}
